// WaveletsMemory_61418032332974
// MI455X (gfx1250) — matrix operand order repaired and checked by reading, not compiled, not run on hardware
//
#include <hip/hip_runtime.h>
#include <hip/hip_bf16.h>
#include <math.h>

typedef __attribute__((ext_vector_type(16))) _Float16 v16h;
typedef __attribute__((ext_vector_type(8)))  _Float16 v8h;
typedef __attribute__((ext_vector_type(8)))  float    v8f;

#define BN_RS 0.99999500003749972f   /* 1/sqrt(1+1e-5) */

// ---------------------------------------------------------------------------
// Encoder: whole 4-layer 1x1-stride2 stack collapses to a per-pixel MLP.
// 65536 pixels (4096 imgs x 16 positions). Weights staged in LDS.
// ---------------------------------------------------------------------------
__global__ void __launch_bounds__(256) k_encoder(
    const float* __restrict__ x,
    const float* __restrict__ w1, const float* __restrict__ b1,
    const float* __restrict__ w2, const float* __restrict__ b2,
    const float* __restrict__ w3, const float* __restrict__ b3,
    const float* __restrict__ w4, const float* __restrict__ b4,
    const float* __restrict__ g1, const float* __restrict__ t1,
    const float* __restrict__ g2, const float* __restrict__ t2,
    const float* __restrict__ g3, const float* __restrict__ t3,
    float* __restrict__ zf32, _Float16* __restrict__ zf16)
{
    __shared__ float sW2[32 * 16];
    __shared__ float sW3[64 * 32];
    __shared__ float sW4[128 * 64];
    __shared__ float sW1[16], sB1[16], sS1[16], sT1[16];
    __shared__ float sB2[32], sS2[32], sT2[32];
    __shared__ float sB3[64], sS3[64], sT3[64];
    __shared__ float sB4[128];

    const int t = threadIdx.x;
    for (int i = t; i < 32 * 16;  i += 256) sW2[i] = w2[i];
    for (int i = t; i < 64 * 32;  i += 256) sW3[i] = w3[i];
    for (int i = t; i < 128 * 64; i += 256) sW4[i] = w4[i];
    if (t < 16)  { sW1[t] = w1[t]; sB1[t] = b1[t]; sS1[t] = g1[t] * BN_RS; sT1[t] = t1[t]; }
    if (t < 32)  { sB2[t] = b2[t]; sS2[t] = g2[t] * BN_RS; sT2[t] = t2[t]; }
    if (t < 64)  { sB3[t] = b3[t]; sS3[t] = g3[t] * BN_RS; sT3[t] = t3[t]; }
    if (t < 128) { sB4[t] = b4[t]; }
    __syncthreads();

    const int gid = blockIdx.x * 256 + t;      // 0..65535
    const int n = gid >> 4;
    const int p = gid & 15;                    // p = yy*4+xx (matches reshape order)
    const float pix = x[(size_t)n * 4096 + (p >> 2) * 1024 + (p & 3) * 16];

    float v1[16];
#pragma unroll
    for (int c = 0; c < 16; ++c)
        v1[c] = fmaxf((sW1[c] * pix + sB1[c]) * sS1[c] + sT1[c], 0.0f);

    float v2[32];
#pragma unroll
    for (int c = 0; c < 32; ++c) {
        float a = sB2[c];
#pragma unroll
        for (int i = 0; i < 16; ++i) a += sW2[c * 16 + i] * v1[i];
        v2[c] = fmaxf(a * sS2[c] + sT2[c], 0.0f);
    }

    float v3[64];
#pragma unroll
    for (int c = 0; c < 64; ++c) {
        float a = sB3[c];
#pragma unroll
        for (int i = 0; i < 32; ++i) a += sW3[c * 32 + i] * v2[i];
        v3[c] = fmaxf(a * sS3[c] + sT3[c], 0.0f);
    }

    const size_t zb = (size_t)n * 2048 + p;
    for (int c = 0; c < 128; ++c) {
        float a = sB4[c];
#pragma unroll
        for (int i = 0; i < 64; ++i) a += sW4[c * 64 + i] * v3[i];
        zf32[zb + (size_t)c * 16] = a;
        zf16[zb + (size_t)c * 16] = (_Float16)a;
    }
}

// ---------------------------------------------------------------------------
// Row L2 norms (block per row).
// ---------------------------------------------------------------------------
__global__ void __launch_bounds__(256) k_rownorm(
    const float* __restrict__ a, float* __restrict__ out, int cols, int validRows)
{
    __shared__ float red[256];
    const int row = blockIdx.x;
    float s = 0.0f;
    if (row < validRows)
        for (int c = threadIdx.x; c < cols; c += 256) {
            float v = a[(size_t)row * cols + c];
            s += v * v;
        }
    red[threadIdx.x] = s;
    __syncthreads();
    for (int off = 128; off > 0; off >>= 1) {
        if (threadIdx.x < off) red[threadIdx.x] += red[threadIdx.x + off];
        __syncthreads();
    }
    if (threadIdx.x == 0) out[row] = (row < validRows) ? sqrtf(red[0]) : 0.0f;
}

// ---------------------------------------------------------------------------
// f16 conversions of memory (row-major, padded to 208 rows) and memory^T
// (2048 x 224, K padded to 224) — makes all WMMA fragment loads contiguous.
// ---------------------------------------------------------------------------
__global__ void __launch_bounds__(256) k_cvt_mem(
    const float* __restrict__ mem, _Float16* __restrict__ mf16)
{
    const int i = blockIdx.x * 256 + threadIdx.x;   // 208*2048
    const int row = i >> 11;
    mf16[i] = (_Float16)((row < 200) ? mem[i] : 0.0f);
}

__global__ void __launch_bounds__(256) k_cvt_memT(
    const float* __restrict__ mem, _Float16* __restrict__ mT16)
{
    const int i = blockIdx.x * 256 + threadIdx.x;   // 2048*224
    const int d = i / 224, m = i % 224;
    mT16[i] = (_Float16)((m < 200) ? mem[(size_t)m * 2048 + d] : 0.0f);
}

// ---------------------------------------------------------------------------
// GEMM1: cosine logits = (z @ mem^T) / max(|z||m|,1e-8).  M=4096 N=208(pad)
// K=2048.  One 16x16 tile per wave, K-chain of v_wmma_f32_16x16x32_f16.
// ---------------------------------------------------------------------------
__global__ void __launch_bounds__(256) k_gemm_logits(
    const _Float16* __restrict__ zf16,   // 4096 x 2048
    const _Float16* __restrict__ mf16,   // 208 x 2048 (zero-padded rows)
    const float* __restrict__ zn,        // 4096
    const float* __restrict__ mn,        // 208 (zero-padded)
    float* __restrict__ logits)          // 4096 x 208
{
    const int lane = threadIdx.x & 31;
    const int wave = threadIdx.x >> 5;
    const int tile = blockIdx.x * 8 + wave;     // 416 blocks * 8 = 3328 = 256*13
    const int tileN = tile % 13;
    const int tileM = tile / 13;

    const int r  = lane & 15;
    const int kh = lane >> 4;

    const _Float16* aRow = zf16 + (size_t)(tileM * 16 + r) * 2048 + kh * 16;
    const _Float16* bCol = mf16 + (size_t)(tileN * 16 + r) * 2048 + kh * 16;

    v8f acc = {};
#pragma unroll 4
    for (int k = 0; k < 2048; k += 32) {
        v8h alo = *(const v8h*)(aRow + k);
        v8h ahi = *(const v8h*)(aRow + k + 8);
        v16h a = __builtin_shufflevector(alo, ahi,
                 0, 1, 2, 3, 4, 5, 6, 7, 8, 9, 10, 11, 12, 13, 14, 15);
        v16h b = *(const v16h*)(bCol + k);
        acc = __builtin_amdgcn_wmma_f32_16x16x32_f16(
                  false, a, false, b, (short)0, acc, false, false);
    }

    const int col = tileN * 16 + r;
    const float mc = mn[col];
    const int rowBase = tileM * 16 + (kh << 3);
#pragma unroll
    for (int i = 0; i < 8; ++i) {
        const int row = rowBase + i;
        const float denom = fmaxf(zn[row] * mc, 1e-8f);
        logits[(size_t)row * 208 + col] = acc[i] / denom;
    }
}

// ---------------------------------------------------------------------------
// Softmax over M=200 + hard shrink + L1 normalize.  One wave per row.
// Writes mem_weight (output #3) and f16 weights padded to K=224 for GEMM2.
// ---------------------------------------------------------------------------
__global__ void __launch_bounds__(256) k_softmax_shrink(
    const float* __restrict__ logits,   // 4096 x 208
    float* __restrict__ mw,             // 4096 x 200 (output)
    _Float16* __restrict__ wf16)        // 4096 x 224
{
    const int lane = threadIdx.x & 31;
    const int wave = threadIdx.x >> 5;
    const int row = blockIdx.x * 8 + wave;
    const float* lr = logits + (size_t)row * 208;

    float vals[7];
    float mx = -1e30f;
#pragma unroll
    for (int tt = 0; tt < 7; ++tt) {
        const int j = lane + tt * 32;
        const float v = (j < 200) ? lr[j] : -1e30f;
        vals[tt] = v;
        mx = fmaxf(mx, v);
    }
    for (int m = 16; m > 0; m >>= 1) mx = fmaxf(mx, __shfl_xor(mx, m, 32));

    float sum = 0.0f;
#pragma unroll
    for (int tt = 0; tt < 7; ++tt) {
        const int j = lane + tt * 32;
        const float e = (j < 200) ? expf(vals[tt] - mx) : 0.0f;
        vals[tt] = e;
        sum += e;
    }
    for (int m = 16; m > 0; m >>= 1) sum += __shfl_xor(sum, m, 32);

    const float thr = 1.0f / 200.0f;
    const float eps = 1e-12f;
    float l1 = 0.0f;
#pragma unroll
    for (int tt = 0; tt < 7; ++tt) {
        const float w = vals[tt] / sum;
        const float d = w - thr;
        const float s = (d > 0.0f) ? (w * d) / (d + eps) : 0.0f;
        vals[tt] = s;
        l1 += s;
    }
    for (int m = 16; m > 0; m >>= 1) l1 += __shfl_xor(l1, m, 32);
    l1 = fmaxf(l1, eps);

#pragma unroll
    for (int tt = 0; tt < 7; ++tt) {
        const int j = lane + tt * 32;          // 0..223
        const float o = vals[tt] / l1;
        if (j < 200) mw[(size_t)row * 200 + j] = o;
        wf16[(size_t)row * 224 + j] = (_Float16)((j < 200) ? o : 0.0f);
    }
}

// ---------------------------------------------------------------------------
// GEMM2: z_hat = mem_weight @ memory.  M=4096 N=2048 K=224(pad, zeros).
// ---------------------------------------------------------------------------
__global__ void __launch_bounds__(256) k_gemm_zhat(
    const _Float16* __restrict__ wf16,   // 4096 x 224
    const _Float16* __restrict__ mT16,   // 2048 x 224 (memory^T, zero-padded K)
    float* __restrict__ zhat)            // 4096 x 2048 (output)
{
    const int lane = threadIdx.x & 31;
    const int wave = threadIdx.x >> 5;
    const int tile = blockIdx.x * 8 + wave;     // 4096 blocks * 8 = 32768 = 256*128
    const int tileN = tile & 127;
    const int tileM = tile >> 7;

    const int r  = lane & 15;
    const int kh = lane >> 4;

    const _Float16* aRow = wf16 + (size_t)(tileM * 16 + r) * 224 + kh * 16;
    const _Float16* bCol = mT16 + (size_t)(tileN * 16 + r) * 224 + kh * 16;

    v8f acc = {};
#pragma unroll
    for (int k = 0; k < 224; k += 32) {
        v8h alo = *(const v8h*)(aRow + k);
        v8h ahi = *(const v8h*)(aRow + k + 8);
        v16h a = __builtin_shufflevector(alo, ahi,
                 0, 1, 2, 3, 4, 5, 6, 7, 8, 9, 10, 11, 12, 13, 14, 15);
        v16h b = *(const v16h*)(bCol + k);
        acc = __builtin_amdgcn_wmma_f32_16x16x32_f16(
                  false, a, false, b, (short)0, acc, false, false);
    }

    const int col = tileN * 16 + r;
    const int rowBase = tileM * 16 + (kh << 3);
#pragma unroll
    for (int i = 0; i < 8; ++i)
        zhat[(size_t)(rowBase + i) * 2048 + col] = acc[i];
}

// ---------------------------------------------------------------------------
// One ConvTranspose2d(k=3,s=2,p=1,op=1) stage, gather form, LDS -> LDS/global.
// Compile-time (Ci,Co,H) so taps unroll fully; weights (Ci,Co,3,3) stream
// from global (L2-resident: same ~0.4MB for every block).
// ---------------------------------------------------------------------------
template <int Ci, int Co, int H, bool RELU>
__device__ __forceinline__ void tconv_stage(
    const float* __restrict__ in,      // LDS, (Ci, H, H)
    float* __restrict__ out,           // LDS or global, (Co, 2H, 2H)
    const float* __restrict__ w,
    const float* __restrict__ bias,
    const float* __restrict__ g,
    const float* __restrict__ t)
{
    constexpr int Ho = 2 * H;
    constexpr int total = Co * Ho * Ho;
    for (int e = threadIdx.x; e < total; e += 256) {
        const int ox = e % Ho;
        const int oy = (e / Ho) % Ho;
        const int o  = e / (Ho * Ho);
        float acc = 0.0f;
#pragma unroll
        for (int ky = 0; ky < 3; ++ky) {
            const int ty = oy + 1 - ky;
            if (ty < 0 || (ty & 1)) continue;
            const int iy = ty >> 1;
            if (iy >= H) continue;
#pragma unroll
            for (int kx = 0; kx < 3; ++kx) {
                const int tx = ox + 1 - kx;
                if (tx < 0 || (tx & 1)) continue;
                const int ix = tx >> 1;
                if (ix >= H) continue;
                const float* ip = in + iy * H + ix;
                const float* wp = w + (o * 3 + ky) * 3 + kx;
                float a = 0.0f;
#pragma unroll 4
                for (int ci = 0; ci < Ci; ++ci)
                    a += ip[ci * (H * H)] * wp[ci * (Co * 9)];
                acc += a;
            }
        }
        float y = acc + bias[o];
        if (RELU) y = fmaxf(y * (g[o] * BN_RS) + t[o], 0.0f);
        out[e] = y;
    }
}

// ---------------------------------------------------------------------------
// Fully fused decoder: one workgroup per image, all intermediates in LDS.
// Ping-pong: bufA(32KB: z -> g2 stage) / bufB(64KB: g1 -> g3 stage).
// HBM traffic: 8KB z read + 16KB recon write per image (vs 220KB unfused).
// ---------------------------------------------------------------------------
__global__ void __launch_bounds__(256) k_decoder_fused(
    const float* __restrict__ zhat,      // (4096, 2048) == (N,128,4,4)
    const float* __restrict__ w0, const float* __restrict__ b0,
    const float* __restrict__ g0, const float* __restrict__ t0,
    const float* __restrict__ w1, const float* __restrict__ b1,
    const float* __restrict__ g1, const float* __restrict__ t1,
    const float* __restrict__ w2, const float* __restrict__ b2,
    const float* __restrict__ g2, const float* __restrict__ t2,
    const float* __restrict__ w3, const float* __restrict__ b3,
    float* __restrict__ recon)           // (4096, 64, 64)
{
    extern __shared__ float lds[];
    float* bufA = lds;           //  8192 floats: z(2048) -> g2(8192) reuse
    float* bufB = lds + 8192;    // 16384 floats: g1(4096) -> g3(16384) reuse

    const int n = blockIdx.x;
    const float* zp = zhat + (size_t)n * 2048;
    for (int i = threadIdx.x; i < 2048; i += 256) bufA[i] = zp[i];
    __syncthreads();

    tconv_stage<128, 64, 4, true >(bufA, bufB, w0, b0, g0, t0);   // ->(64,8,8)
    __syncthreads();
    tconv_stage< 64, 32, 8, true >(bufB, bufA, w1, b1, g1, t1);   // ->(32,16,16)
    __syncthreads();
    tconv_stage< 32, 16, 16, true >(bufA, bufB, w2, b2, g2, t2);  // ->(16,32,32)
    __syncthreads();
    tconv_stage< 16, 1, 32, false>(bufB, recon + (size_t)n * 4096,
                                   w3, b3, b3, b3);               // ->(1,64,64)
}

// ---------------------------------------------------------------------------
// Host-side launcher
// ---------------------------------------------------------------------------
extern "C" void kernel_launch(void* const* d_in, const int* in_sizes, int n_in,
                              void* d_out, int out_size, void* d_ws, size_t ws_size,
                              hipStream_t stream)
{
    (void)in_sizes; (void)n_in; (void)out_size; (void)ws_size;

    const float* x   = (const float*)d_in[0];
    const float* mem = (const float*)d_in[1];
    const float* enc_w[4] = {(const float*)d_in[2],  (const float*)d_in[3],
                             (const float*)d_in[4],  (const float*)d_in[5]};
    const float* enc_b[4] = {(const float*)d_in[6],  (const float*)d_in[7],
                             (const float*)d_in[8],  (const float*)d_in[9]};
    const float* enc_g[3] = {(const float*)d_in[10], (const float*)d_in[11],
                             (const float*)d_in[12]};
    const float* enc_t[3] = {(const float*)d_in[13], (const float*)d_in[14],
                             (const float*)d_in[15]};
    const float* dec_w[4] = {(const float*)d_in[16], (const float*)d_in[17],
                             (const float*)d_in[18], (const float*)d_in[19]};
    const float* dec_b[4] = {(const float*)d_in[20], (const float*)d_in[21],
                             (const float*)d_in[22], (const float*)d_in[23]};
    const float* dec_g[3] = {(const float*)d_in[24], (const float*)d_in[25],
                             (const float*)d_in[26]};
    const float* dec_t[3] = {(const float*)d_in[27], (const float*)d_in[28],
                             (const float*)d_in[29]};

    // Output regions: x_recon | z_hat | mem_weight
    float* out_recon = (float*)d_out;
    float* out_zhat  = (float*)d_out + (size_t)4096 * 64 * 64;   // +16,777,216
    float* out_mw    = out_zhat + (size_t)4096 * 2048;           // +8,388,608

    // Workspace carve-up (256B aligned)
    char*  ws  = (char*)d_ws;
    size_t off = 0;
    auto carve = [&](size_t bytes) -> char* {
        char* p = ws + off;
        off = (off + bytes + 255) & ~(size_t)255;
        return p;
    };
    float*     z_f32  = (float*)    carve((size_t)4096 * 2048 * 4);
    _Float16*  z_f16  = (_Float16*) carve((size_t)4096 * 2048 * 2);
    float*     zn     = (float*)    carve((size_t)4096 * 4);
    float*     mn     = (float*)    carve((size_t)208 * 4);
    _Float16*  m_f16  = (_Float16*) carve((size_t)208 * 2048 * 2);
    _Float16*  mT_f16 = (_Float16*) carve((size_t)2048 * 224 * 2);
    float*     logits = (float*)    carve((size_t)4096 * 208 * 4);
    _Float16*  w_f16  = (_Float16*) carve((size_t)4096 * 224 * 2);

    // 1) Encoder (collapsed MLP) -> z (f32 + f16)
    k_encoder<<<256, 256, 0, stream>>>(
        x, enc_w[0], enc_b[0], enc_w[1], enc_b[1], enc_w[2], enc_b[2],
        enc_w[3], enc_b[3], enc_g[0], enc_t[0], enc_g[1], enc_t[1],
        enc_g[2], enc_t[2], z_f32, z_f16);

    // 2) Row norms
    k_rownorm<<<4096, 256, 0, stream>>>(z_f32, zn, 2048, 4096);
    k_rownorm<<<208,  256, 0, stream>>>(mem,   mn, 2048, 200);

    // 3) f16 conversions of memory (+ transpose, zero padding)
    k_cvt_mem <<<(208 * 2048) / 256, 256, 0, stream>>>(mem, m_f16);
    k_cvt_memT<<<(2048 * 224) / 256, 256, 0, stream>>>(mem, mT_f16);

    // 4) Cosine-similarity GEMM (WMMA)
    k_gemm_logits<<<416, 256, 0, stream>>>(z_f16, m_f16, zn, mn, logits);

    // 5) Softmax + hard shrink + L1 normalize -> mem_weight (out) + w_f16
    k_softmax_shrink<<<512, 256, 0, stream>>>(logits, out_mw, w_f16);

    // 6) z_hat = mem_weight @ memory (WMMA) -> output, reused as decoder input
    k_gemm_zhat<<<4096, 256, 0, stream>>>(w_f16, mT_f16, out_zhat);

    // 7) Fully fused decoder (one block per image, 96KB dynamic LDS)
    k_decoder_fused<<<4096, 256, 24576 * sizeof(float), stream>>>(
        out_zhat,
        dec_w[0], dec_b[0], dec_g[0], dec_t[0],
        dec_w[1], dec_b[1], dec_g[1], dec_t[1],
        dec_w[2], dec_b[2], dec_g[2], dec_t[2],
        dec_w[3], dec_b[3],
        out_recon);
}
